// st_parallel_attention_32925219291321
// MI455X (gfx1250) — compile-verified, matrix operand order checked by reading, not run on hardware
//
#include <hip/hip_runtime.h>

#define B_ 4
#define S_ 64
#define N_ 256
#define D_ 512
#define H_ 8
#define DEPTH_ 64

typedef __attribute__((ext_vector_type(16))) __bf16  bf16x16;
typedef __attribute__((ext_vector_type(8)))  float   f32x8;

union ABFrag { bf16x16 v; uint4 q[2]; };

__device__ __forceinline__ unsigned short f2bf(float f) {
    union { float f; unsigned u; } x; x.f = f;
    unsigned u = x.u;
    // round-to-nearest-even truncation to bf16
    return (unsigned short)((u + 0x7FFFu + ((u >> 16) & 1u)) >> 16);
}

// ---------------- Kernel 1a: x (f32) -> bf16 bits ----------------
__global__ void cvt_x_kernel(const float* __restrict__ x,
                             unsigned short* __restrict__ xb, int n) {
    int i = blockIdx.x * blockDim.x + threadIdx.x;
    if (i < n) xb[i] = f2bf(x[i]);
}

// ------------- Kernel 1b: W (512x512 f32) -> bf16 transposed -------------
__global__ void cvt_wt_kernel(const float* __restrict__ w,
                              unsigned short* __restrict__ wt) {
    int i = blockIdx.x * blockDim.x + threadIdx.x;   // over 512*512
    int row = i >> 9, col = i & 511;
    wt[col * D_ + row] = f2bf(w[i]);                  // wt[j][i] = w[i][j]
}

// ------------- Kernel: row-block matvec  out[r][j] = X[r,:]·W[:,j] + b[j] -------------
// 256 blocks (rows), 512 threads (cols). Used for q-projection and output projection.
__global__ void matvec512_kernel(const float* __restrict__ X, int xstride,
                                 const float* __restrict__ W,
                                 const float* __restrict__ bias,
                                 float* __restrict__ out) {
    const int r = blockIdx.x;
    const int j = threadIdx.x;
    const float* xr = X + (size_t)r * xstride;
    float acc = bias[j];
    #pragma unroll 8
    for (int i = 0; i < D_; ++i)
        acc = fmaf(xr[i], W[i * D_ + j], acc);
    out[(size_t)r * D_ + j] = acc;
}

// ------------- WMMA tile: C(256x64) = x_bs(256x512) * WT[h-slice]^T + bias -------------
// Per-wave: rows [rowBase, rowBase+32), all 64 cols -> 2x4 accumulators.
__device__ __forceinline__ void compute_tile(
    const unsigned short* __restrict__ xbase,   // bf16 bits, (N_, D_) row-major
    const unsigned short* __restrict__ WT,      // bf16 bits, (D_, D_), row j = output col
    const float* __restrict__ bias,
    int h, int rowBase, int half, int nloc,
    float* __restrict__ tile)                   // LDS, stride 65
{
    f32x8 acc[2][4];
    const f32x8 zero = {0.f, 0.f, 0.f, 0.f, 0.f, 0.f, 0.f, 0.f};
    #pragma unroll
    for (int rt = 0; rt < 2; ++rt)
        #pragma unroll
        for (int ct = 0; ct < 4; ++ct) acc[rt][ct] = zero;

    const unsigned short* xr0 = xbase + (size_t)(rowBase + nloc)      * D_;
    const unsigned short* xr1 = xbase + (size_t)(rowBase + 16 + nloc) * D_;
    const unsigned short* wr[4];
    #pragma unroll
    for (int ct = 0; ct < 4; ++ct)
        wr[ct] = WT + (size_t)(h * DEPTH_ + ct * 16 + nloc) * D_;

    #pragma unroll 2
    for (int kk = 0; kk < D_; kk += 32) {
        const int k0 = kk + half * 8;             // A/B frag: K = k0..k0+7, k0+16..k0+23
        ABFrag a0, a1, b[4];
        a0.q[0] = *(const uint4*)(xr0 + k0);
        a0.q[1] = *(const uint4*)(xr0 + k0 + 16);
        a1.q[0] = *(const uint4*)(xr1 + k0);
        a1.q[1] = *(const uint4*)(xr1 + k0 + 16);
        #pragma unroll
        for (int ct = 0; ct < 4; ++ct) {
            b[ct].q[0] = *(const uint4*)(wr[ct] + k0);
            b[ct].q[1] = *(const uint4*)(wr[ct] + k0 + 16);
        }
        #pragma unroll
        for (int ct = 0; ct < 4; ++ct) {
            acc[0][ct] = __builtin_amdgcn_wmma_f32_16x16x32_bf16(
                false, a0.v, false, b[ct].v, (short)0, acc[0][ct], false, false);
            acc[1][ct] = __builtin_amdgcn_wmma_f32_16x16x32_bf16(
                false, a1.v, false, b[ct].v, (short)0, acc[1][ct], false, false);
        }
    }

    // C layout: lane n = nloc, VGPR r -> M = r + 8*half
    #pragma unroll
    for (int rt = 0; rt < 2; ++rt) {
        #pragma unroll
        for (int ct = 0; ct < 4; ++ct) {
            const int col = ct * 16 + nloc;
            const float bb = bias[h * DEPTH_ + col];
            #pragma unroll
            for (int r = 0; r < 8; ++r) {
                const int m = rowBase + rt * 16 + r + 8 * half;
                tile[m * 65 + col] = acc[rt][ct][r] + bb;
            }
        }
    }
}

// ------------- Kernel 2: fused per-(b,s,h) projection + attention -------------
__launch_bounds__(256)
__global__ void attn_kernel(const unsigned short* __restrict__ xb,   // (B*S, N, D) bf16
                            const unsigned short* __restrict__ WkT,
                            const float* __restrict__ bk,
                            const unsigned short* __restrict__ WvT,
                            const float* __restrict__ bv,
                            const float* __restrict__ q_ws,          // (B*S, D) f32
                            float* __restrict__ concat_ws)           // (B*S, D) f32
{
    __shared__ float tile[N_ * 65];   // K-slice then V-slice (f32), 64-bank-padded
    __shared__ float attnw[N_];
    __shared__ float red[N_];
    __shared__ float qs[DEPTH_];

    const int tid  = threadIdx.x;
    const int wave = tid >> 5;
    const int lane = tid & 31;
    const int half = lane >> 4;
    const int nloc = lane & 15;
    const int bs   = blockIdx.x / H_;
    const int h    = blockIdx.x % H_;
    const int rowBase = wave * 32;

    const unsigned short* xbase = xb + (size_t)bs * (N_ * D_);

    if (tid < DEPTH_) qs[tid] = q_ws[(size_t)bs * D_ + h * DEPTH_ + tid];

    // ---- K-slice via WMMA ----
    compute_tile(xbase, WkT, bk, h, rowBase, half, nloc, tile);
    __syncthreads();

    // logits[n] = (q_h · k_n) / sqrt(64)
    {
        const float* kr = &tile[tid * 65];
        float acc = 0.f;
        #pragma unroll 16
        for (int d = 0; d < DEPTH_; ++d) acc = fmaf(qs[d], kr[d], acc);
        attnw[tid] = acc * 0.125f;
        red[tid]   = attnw[tid];
    }
    __syncthreads();

    // max reduction (256 -> 1)
    #pragma unroll
    for (int off = 128; off > 0; off >>= 1) {
        if (tid < off) red[tid] = fmaxf(red[tid], red[tid + off]);
        __syncthreads();
    }
    const float mx = red[0];
    __syncthreads();

    const float e = __expf(attnw[tid] - mx);
    attnw[tid] = e;
    red[tid]   = e;
    __syncthreads();

    // sum reduction
    #pragma unroll
    for (int off = 128; off > 0; off >>= 1) {
        if (tid < off) red[tid] += red[tid + off];
        __syncthreads();
    }
    const float inv_sum = 1.0f / red[0];
    __syncthreads();

    // ---- V-slice via WMMA (reuses tile) ----
    compute_tile(xbase, WvT, bv, h, rowBase, half, nloc, tile);
    __syncthreads();

    // ctx[d] = (1/sum) * sum_n attnw[n] * v[n][d]; 4 partial chunks over n
    {
        const int d = tid & 63;
        const int chunk = tid >> 6;
        float p = 0.f;
        #pragma unroll 8
        for (int n = chunk * 64; n < chunk * 64 + 64; ++n)
            p = fmaf(attnw[n], tile[n * 65 + d], p);
        red[tid] = p;
    }
    __syncthreads();
    if (tid < DEPTH_) {
        const float ctx = (red[tid] + red[64 + tid] + red[128 + tid] + red[192 + tid]) * inv_sum;
        // concat[b,s,0,h*64+d] = ctx[b,h,s,0,d]
        concat_ws[(size_t)bs * D_ + h * DEPTH_ + tid] = ctx;
    }
}

extern "C" void kernel_launch(void* const* d_in, const int* in_sizes, int n_in,
                              void* d_out, int out_size, void* d_ws, size_t ws_size,
                              hipStream_t stream) {
    const float* x  = (const float*)d_in[0];
    const float* Wq = (const float*)d_in[1];
    const float* bq = (const float*)d_in[2];
    const float* Wk = (const float*)d_in[3];
    const float* bk = (const float*)d_in[4];
    const float* Wv = (const float*)d_in[5];
    const float* bv = (const float*)d_in[6];
    const float* Wo = (const float*)d_in[7];
    const float* bo = (const float*)d_in[8];
    float* out = (float*)d_out;

    const int n_x = B_ * S_ * N_ * D_;   // 33,554,432
    const int n_w = D_ * D_;             // 262,144
    const int n_r = B_ * S_;             // 256 rows for q / output projections

    // workspace carve-up (~66 MiB)
    char* ws = (char*)d_ws;
    unsigned short* xb  = (unsigned short*)ws;                         // 64 MiB
    unsigned short* wkT = (unsigned short*)(ws + (size_t)n_x * 2);     // 512 KiB
    unsigned short* wvT = wkT + n_w;                                   // 512 KiB
    float* q_ws  = (float*)((char*)(wvT + n_w));                       // 512 KiB
    float* cc_ws = q_ws + (size_t)n_r * D_;                            // 512 KiB

    // 1) convert x to bf16; transpose+convert Wk, Wv
    cvt_x_kernel<<<n_x / 256, 256, 0, stream>>>(x, xb, n_x);
    cvt_wt_kernel<<<n_w / 256, 256, 0, stream>>>(Wk, wkT);
    cvt_wt_kernel<<<n_w / 256, 256, 0, stream>>>(Wv, wvT);

    // 2) q projection from cls token rows: x[b,s,0,:] (stride N_*D_ between rows)
    matvec512_kernel<<<n_r, D_, 0, stream>>>(x, N_ * D_, Wq, bq, q_ws);

    // 3) fused per-(b,s,h) K/V projection + attention (WMMA path)
    attn_kernel<<<B_ * S_ * H_, 256, 0, stream>>>(xb, wkT, bk, wvT, bv, q_ws, cc_ws);

    // 4) output projection
    matvec512_kernel<<<n_r, D_, 0, stream>>>(cc_ws, D_, Wo, bo, out);
}
